// EfficientSelfAttention_1692217115081
// MI455X (gfx1250) — matrix operand order repaired and checked by reading, not compiled, not run on hardware
//
#include <hip/hip_runtime.h>

#define DEV __device__ __forceinline__

typedef __attribute__((ext_vector_type(16))) __bf16 v16bf;
typedef __attribute__((ext_vector_type(8)))  float  v8f;

union BFrag { v16bf v; __bf16 e[16]; };
union FAcc8 { v8f v; float e[8]; };

DEV int aKidx(int j, int lane) { return j + 8 * ((j >> 3) + (lane >> 4)); }
DEV int bKidx(int j, int lane) { return j + 8 * ((j >> 3) + (lane >> 4)); }

// native bf16 convert (gfx1250 has hardware cvt; avoids 4-VALU software RNE)
DEV __bf16 f2bf(float f) { return (__bf16)f; }

constexpr int BATCH = 8, NTOK = 16384, CH = 64, HW = 128, NR = 256;
constexpr float SCALE = 0.125f; // 64^-0.5

// workspace offsets, in __bf16 elements
constexpr size_t QWF_OFF     = 0;       //  4*2*32*16   =   4096
constexpr size_t PWF_OFF     = 4096;    //                  4096
constexpr size_t KVWF_OFF    = 8192;    //  8*2*32*16   =   8192
constexpr size_t SRWF_OFF    = 16384;   //  4*128*32*16 = 262144
constexpr size_t KVPLAIN_OFF = 278528;  //  8*256*128   = 262144
constexpr size_t KTF_OFF     = 540672;  //  8*16*2*32*16= 131072
constexpr size_t VF_OFF      = 671744;  //  8*4*8*32*16 = 131072
// total 802816 bf16 = 1,605,632 bytes

// ---------------------------------------------------------------------------
// Kernel P: repack weights into WMMA B-fragment layout (bf16).
// For the conv GEMM the K dimension is reordered as e = (ky*8+kx)*64 + c so
// that the activation gather in kv_path reads contiguous channel runs.
// ---------------------------------------------------------------------------
__global__ __launch_bounds__(256) void prep_weights(
    const float* __restrict__ qw, const float* __restrict__ pw,
    const float* __restrict__ kvw, const float* __restrict__ srw,
    __bf16* __restrict__ ws) {
  int tid  = blockIdx.x * 256 + threadIdx.x;
  int lane = tid & 31;
  int g    = tid >> 5;
  BFrag f;
  if (g < 512) {                       // sr_w: (64 out, K=4096), nt in [0,4), kc in [0,128)
    int nt = g >> 7, kc = g & 127;
    int n = nt * 16 + (lane & 15);
    #pragma unroll
    for (int j = 0; j < 16; ++j) {
      int k = kc * 32 + bKidx(j, lane);          // reordered K index
      int c = k & 63, s = k >> 6;                // s = ky*8+kx
      f.e[j] = f2bf(srw[(size_t)n * 4096 + c * 64 + s]);
    }
    *(v16bf*)(ws + SRWF_OFF + ((size_t)g * 32 + lane) * 16) = f.v;
  } else if (g < 520) {                // q_w: (64,64), nt in [0,4), kc in [0,2)
    int gq = g - 512, nt = gq >> 1, kc = gq & 1;
    int n = nt * 16 + (lane & 15);
    #pragma unroll
    for (int j = 0; j < 16; ++j) f.e[j] = f2bf(qw[n * 64 + kc * 32 + bKidx(j, lane)]);
    *(v16bf*)(ws + QWF_OFF + ((size_t)gq * 32 + lane) * 16) = f.v;
  } else if (g < 528) {                // proj_w
    int gp = g - 520, nt = gp >> 1, kc = gp & 1;
    int n = nt * 16 + (lane & 15);
    #pragma unroll
    for (int j = 0; j < 16; ++j) f.e[j] = f2bf(pw[n * 64 + kc * 32 + bKidx(j, lane)]);
    *(v16bf*)(ws + PWF_OFF + ((size_t)gp * 32 + lane) * 16) = f.v;
  } else if (g < 544) {                // kv_w: (128,64), nt in [0,8), kc in [0,2)
    int gk = g - 528, nt = gk >> 1, kc = gk & 1;
    int n = nt * 16 + (lane & 15);
    #pragma unroll
    for (int j = 0; j < 16; ++j) f.e[j] = f2bf(kvw[n * 64 + kc * 32 + bKidx(j, lane)]);
    *(v16bf*)(ws + KVWF_OFF + ((size_t)gk * 32 + lane) * 16) = f.v;
  }
}

// ---------------------------------------------------------------------------
// Kernel 1: conv(8x8,s8) as GEMM (M=256,K=4096,N=64) -> +bias -> LayerNorm
//           -> kv linear (N=128) -> plain bf16 kv in ws.  One block per batch.
// ---------------------------------------------------------------------------
__global__ __launch_bounds__(256) void kv_path(
    const float* __restrict__ x, const float* __restrict__ sr_b,
    const float* __restrict__ ln_g, const float* __restrict__ ln_b,
    const float* __restrict__ kv_b, const __bf16* __restrict__ ws_ro,
    __bf16* __restrict__ ws) {
  __shared__ float convbuf[NR * CH];   // 64 KB
  const __bf16* srwf = ws_ro + SRWF_OFF;
  const __bf16* kvwf = ws_ro + KVWF_OFF;
  __bf16* kvplain = ws + KVPLAIN_OFF;

  int b = blockIdx.x;
  int lane = threadIdx.x & 31, w = threadIdx.x >> 5;
  int m16 = lane & 15, hi = lane >> 4;
  const float* xb = x + (size_t)b * NTOK * CH;

  // --- conv as GEMM, 2 M-tiles per wave; K reordered as (spatial, channel) ---
  for (int mt = w * 2; mt < w * 2 + 2; ++mt) {
    FAcc8 acc[4];
    #pragma unroll
    for (int nt = 0; nt < 4; ++nt) acc[nt].v = (v8f)(0.0f);
    int tr = mt * 16 + m16, py = tr >> 4, px = tr & 15;
    for (int kc = 0; kc < 128; ++kc) {
      int s = kc >> 1;                 // spatial index ky*8+kx (fixed per chunk)
      int ky = s >> 3, kx = s & 7;
      const float* xrow = xb + ((size_t)((py * 8 + ky) * HW + px * 8 + kx)) * CH
                             + (kc & 1) * 32;   // 32 contiguous channels
      BFrag a;
      #pragma unroll
      for (int j = 0; j < 16; ++j)
        a.e[j] = f2bf(xrow[aKidx(j, lane)]);    // two contiguous 8-float runs
      #pragma unroll
      for (int nt = 0; nt < 4; ++nt) {
        BFrag bb;
        bb.v = *(const v16bf*)(srwf + (((size_t)nt * 128 + kc) * 32 + lane) * 16);
        acc[nt].v = __builtin_amdgcn_wmma_f32_16x16x32_bf16(
            false, a.v, false, bb.v, (short)0, acc[nt].v, false, false);
      }
    }
    #pragma unroll
    for (int nt = 0; nt < 4; ++nt) {
      int n = nt * 16 + m16;
      float bias = sr_b[n];
      #pragma unroll
      for (int v = 0; v < 8; ++v) {
        int row = mt * 16 + v + 8 * hi;
        convbuf[row * CH + n] = acc[nt].e[v] + bias;
      }
    }
  }
  __syncthreads();

  // --- LayerNorm: one token per thread, rewrite row in place as bf16 ---
  {
    int t = threadIdx.x;
    float r[64];
    float mu = 0.f;
    #pragma unroll
    for (int c = 0; c < 64; ++c) { r[c] = convbuf[t * 64 + c]; mu += r[c]; }
    mu *= (1.f / 64.f);
    float var = 0.f;
    #pragma unroll
    for (int c = 0; c < 64; ++c) { float d = r[c] - mu; var += d * d; }
    var *= (1.f / 64.f);
    float rs = rsqrtf(var + 1e-5f);
    __bf16* xn = (__bf16*)(convbuf + t * 64);   // row-private bytes
    #pragma unroll
    for (int c = 0; c < 64; ++c) xn[c] = f2bf((r[c] - mu) * rs * ln_g[c] + ln_b[c]);
  }
  __syncthreads();

  // --- kv GEMM: M=256, K=64, N=128 ---
  for (int mt = w * 2; mt < w * 2 + 2; ++mt) {
    BFrag a[2];
    #pragma unroll
    for (int kc = 0; kc < 2; ++kc)
      #pragma unroll
      for (int j = 0; j < 16; ++j) {
        int row = mt * 16 + m16;
        a[kc].e[j] = ((const __bf16*)(convbuf + row * 64))[kc * 32 + aKidx(j, lane)];
      }
    #pragma unroll
    for (int nt = 0; nt < 8; ++nt) {
      FAcc8 acc; acc.v = (v8f)(0.0f);
      #pragma unroll
      for (int kc = 0; kc < 2; ++kc) {
        BFrag bb;
        bb.v = *(const v16bf*)(kvwf + (((size_t)nt * 2 + kc) * 32 + lane) * 16);
        acc.v = __builtin_amdgcn_wmma_f32_16x16x32_bf16(
            false, a[kc].v, false, bb.v, (short)0, acc.v, false, false);
      }
      int n = nt * 16 + m16;
      float bias = kv_b[n];
      #pragma unroll
      for (int v = 0; v < 8; ++v) {
        int row = mt * 16 + v + 8 * hi;
        kvplain[((size_t)b * NR + row) * 128 + n] = f2bf(acc.e[v] + bias);
      }
    }
  }
}

// ---------------------------------------------------------------------------
// Kernel S: repack K (B-matrix of Q.K^T, Kdim=64) and V (B-matrix of P.V,
// Kdim=256 tokens) into fragment-ready bf16 layouts.
// ---------------------------------------------------------------------------
__global__ __launch_bounds__(256) void swizzle_kv(__bf16* __restrict__ ws) {
  const __bf16* kvplain = ws + KVPLAIN_OFF;
  int tid = blockIdx.x * 256 + threadIdx.x;
  int lane = tid & 31, g = tid >> 5;
  BFrag f;
  if (g < 256) {                       // ktfrag: g = b*32 + nt*2 + kc ; nt in [0,16)
    int b = g >> 5, r = g & 31, nt = r >> 1, kc = r & 1;
    int ntok = nt * 16 + (lane & 15);  // column = reduced token
    #pragma unroll
    for (int j = 0; j < 16; ++j) {
      int kd = kc * 32 + bKidx(j, lane);       // head-dim index (<64)
      f.e[j] = kvplain[((size_t)b * NR + ntok) * 128 + kd];
    }
    *(v16bf*)(ws + KTF_OFF + ((size_t)g * 32 + lane) * 16) = f.v;
  } else {                             // vfrag: g2 = b*32 + nt*8 + kc ; nt in [0,4)
    int g2 = g - 256;
    int b = g2 >> 5, r = g2 & 31, nt = r >> 3, kc = r & 7;
    int n = nt * 16 + (lane & 15);     // column = head dim
    #pragma unroll
    for (int j = 0; j < 16; ++j) {
      int ktok = kc * 32 + bKidx(j, lane);     // reduced-token index (<256)
      f.e[j] = kvplain[((size_t)b * NR + ktok) * 128 + 64 + n];
    }
    *(v16bf*)(ws + VF_OFF + ((size_t)g2 * 32 + lane) * 16) = f.v;
  }
}

// ---------------------------------------------------------------------------
// Kernel 2: fused q-proj -> Q.K^T -> softmax -> P.V -> out-proj.
// 1024 blocks (8 batch x 128 tiles of 128 tokens), 8 waves x 16-token M-tile.
// x tile is staged into LDS with ASYNCcnt-tracked async loads.
// ---------------------------------------------------------------------------
__global__ __launch_bounds__(256) void attention(
    const float* __restrict__ x, const float* __restrict__ q_b,
    const float* __restrict__ proj_b, const __bf16* __restrict__ ws,
    float* __restrict__ out) {
  __shared__ __align__(32) __bf16 smem[32768];   // 64 KB, phase-unioned
  const __bf16* qwf = ws + QWF_OFF;
  const __bf16* pwf = ws + PWF_OFF;
  const __bf16* ktf = ws + KTF_OFF;
  const __bf16* vf  = ws + VF_OFF;

  int b = blockIdx.x >> 7;
  int tile0 = (blockIdx.x & 127) * 128;
  int lane = threadIdx.x & 31, w = threadIdx.x >> 5;
  int m16 = lane & 15, hi = lane >> 4;

  // --- phase 0: async-copy x tile (128 rows x 64 ch f32 = 32 KB) into LDS ---
  {
    unsigned lds_base =
        (unsigned)(unsigned long long)(__attribute__((address_space(3))) char*)smem;
    const char* gsrc = (const char*)(x + ((size_t)b * NTOK + tile0) * CH);
    #pragma unroll
    for (int i = 0; i < 8; ++i) {
      unsigned off = (unsigned)(w * 4096 + i * 512 + lane * 16);
      unsigned ldsaddr = lds_base + off;
      unsigned long long gaddr = (unsigned long long)(gsrc + off);
      asm volatile("global_load_async_to_lds_b128 %0, %1, off"
                   :: "v"(ldsaddr), "v"(gaddr) : "memory");
    }
    asm volatile("s_wait_asynccnt 0" ::: "memory");
  }
  __syncthreads();

  // --- phase 1: q = x @ q_w^T + q_b (A fragments from LDS x tile) ---
  const float* xt = (const float*)smem;          // 128 x 64 f32
  BFrag ax[2];
  #pragma unroll
  for (int kc = 0; kc < 2; ++kc)
    #pragma unroll
    for (int j = 0; j < 16; ++j)
      ax[kc].e[j] = f2bf(xt[(w * 16 + m16) * 64 + kc * 32 + aKidx(j, lane)]);
  __syncthreads();                               // x reads done before qbuf overlay

  FAcc8 accq[4];
  #pragma unroll
  for (int nt = 0; nt < 4; ++nt) accq[nt].v = (v8f)(0.0f);
  #pragma unroll
  for (int kc = 0; kc < 2; ++kc)
    #pragma unroll
    for (int nt = 0; nt < 4; ++nt) {
      BFrag bb;
      bb.v = *(const v16bf*)(qwf + (((size_t)nt * 2 + kc) * 32 + lane) * 16);
      accq[nt].v = __builtin_amdgcn_wmma_f32_16x16x32_bf16(
          false, ax[kc].v, false, bb.v, (short)0, accq[nt].v, false, false);
    }
  // stage q (D-layout) into LDS rows, re-read as A-fragments
  #pragma unroll
  for (int nt = 0; nt < 4; ++nt) {
    int n = nt * 16 + m16;
    float bias = q_b[n];
    #pragma unroll
    for (int v = 0; v < 8; ++v)
      smem[(w * 16 + v + 8 * hi) * 64 + n] = f2bf(accq[nt].e[v] + bias);
  }
  __syncthreads();
  BFrag aq[2];
  #pragma unroll
  for (int kc = 0; kc < 2; ++kc)
    #pragma unroll
    for (int j = 0; j < 16; ++j)
      aq[kc].e[j] = smem[(w * 16 + m16) * 64 + kc * 32 + aKidx(j, lane)];
  __syncthreads();   // everyone done with qbuf before p overlays it

  // --- phase 2: S = Q @ K^T  (16 x 256 per wave) ---
  FAcc8 s[16];
  #pragma unroll
  for (int nt = 0; nt < 16; ++nt) s[nt].v = (v8f)(0.0f);
  #pragma unroll
  for (int kc = 0; kc < 2; ++kc)
    #pragma unroll
    for (int nt = 0; nt < 16; ++nt) {
      BFrag kb;
      kb.v = *(const v16bf*)(ktf + ((((size_t)b * 16 + nt) * 2 + kc) * 32 + lane) * 16);
      s[nt].v = __builtin_amdgcn_wmma_f32_16x16x32_bf16(
          false, aq[kc].v, false, kb.v, (short)0, s[nt].v, false, false);
    }

  // warm V fragments into cache while softmax runs
  __builtin_prefetch(vf + (((size_t)b * 4) * 8 * 32 + lane) * 16, 0, 3);

  // --- softmax over 256 columns: rows live across 16-lane half-waves ---
  float rinv[8];
  #pragma unroll
  for (int v = 0; v < 8; ++v) {
    float mx = -3.0e38f;
    #pragma unroll
    for (int nt = 0; nt < 16; ++nt) {
      s[nt].e[v] *= SCALE;
      mx = fmaxf(mx, s[nt].e[v]);
    }
    mx = fmaxf(mx, __shfl_xor(mx, 1, 32));
    mx = fmaxf(mx, __shfl_xor(mx, 2, 32));
    mx = fmaxf(mx, __shfl_xor(mx, 4, 32));
    mx = fmaxf(mx, __shfl_xor(mx, 8, 32));
    float sum = 0.f;
    #pragma unroll
    for (int nt = 0; nt < 16; ++nt) {
      float e = __expf(s[nt].e[v] - mx);
      s[nt].e[v] = e;
      sum += e;
    }
    sum += __shfl_xor(sum, 1, 32);
    sum += __shfl_xor(sum, 2, 32);
    sum += __shfl_xor(sum, 4, 32);
    sum += __shfl_xor(sum, 8, 32);
    rinv[v] = 1.f / sum;
  }

  // stage P (bf16) per-wave, re-read as A-fragments
  __bf16* pbuf = smem + w * 4096;    // 16 x 256
  #pragma unroll
  for (int nt = 0; nt < 16; ++nt)
    #pragma unroll
    for (int v = 0; v < 8; ++v)
      pbuf[(v + 8 * hi) * 256 + nt * 16 + m16] = f2bf(s[nt].e[v] * rinv[v]);
  __syncthreads();

  // --- phase 3: O = P @ V ---
  FAcc8 acco[4];
  #pragma unroll
  for (int nt = 0; nt < 4; ++nt) acco[nt].v = (v8f)(0.0f);
  #pragma unroll
  for (int kc = 0; kc < 8; ++kc) {
    BFrag ap;
    #pragma unroll
    for (int j = 0; j < 16; ++j)
      ap.e[j] = pbuf[m16 * 256 + kc * 32 + aKidx(j, lane)];
    #pragma unroll
    for (int nt = 0; nt < 4; ++nt) {
      BFrag vb;
      vb.v = *(const v16bf*)(vf + ((((size_t)b * 4 + nt) * 8 + kc) * 32 + lane) * 16);
      acco[nt].v = __builtin_amdgcn_wmma_f32_16x16x32_bf16(
          false, ap.v, false, vb.v, (short)0, acco[nt].v, false, false);
    }
  }
  __syncthreads();   // p reads done before obuf overlays

  // stage O, re-read as A-fragments
  #pragma unroll
  for (int nt = 0; nt < 4; ++nt)
    #pragma unroll
    for (int v = 0; v < 8; ++v)
      smem[(w * 16 + v + 8 * hi) * 64 + nt * 16 + m16] = f2bf(acco[nt].e[v]);
  __syncthreads();
  BFrag ao[2];
  #pragma unroll
  for (int kc = 0; kc < 2; ++kc)
    #pragma unroll
    for (int j = 0; j < 16; ++j)
      ao[kc].e[j] = smem[(w * 16 + m16) * 64 + kc * 32 + aKidx(j, lane)];

  // --- phase 4: out = O @ proj_w^T + proj_b ---
  FAcc8 y[4];
  #pragma unroll
  for (int nt = 0; nt < 4; ++nt) y[nt].v = (v8f)(0.0f);
  #pragma unroll
  for (int kc = 0; kc < 2; ++kc)
    #pragma unroll
    for (int nt = 0; nt < 4; ++nt) {
      BFrag bb;
      bb.v = *(const v16bf*)(pwf + (((size_t)nt * 2 + kc) * 32 + lane) * 16);
      y[nt].v = __builtin_amdgcn_wmma_f32_16x16x32_bf16(
          false, ao[kc].v, false, bb.v, (short)0, y[nt].v, false, false);
    }
  #pragma unroll
  for (int nt = 0; nt < 4; ++nt) {
    int n = nt * 16 + m16;
    float bias = proj_b[n];
    #pragma unroll
    for (int v = 0; v < 8; ++v) {
      int token = tile0 + w * 16 + v + 8 * hi;
      out[((size_t)b * NTOK + token) * 64 + n] = y[nt].e[v] + bias;
    }
  }
}

// ---------------------------------------------------------------------------
extern "C" void kernel_launch(void* const* d_in, const int* in_sizes, int n_in,
                              void* d_out, int out_size, void* d_ws, size_t ws_size,
                              hipStream_t stream) {
  const float* x      = (const float*)d_in[0];
  const float* sr_w   = (const float*)d_in[1];
  const float* sr_b   = (const float*)d_in[2];
  const float* ln_g   = (const float*)d_in[3];
  const float* ln_b   = (const float*)d_in[4];
  const float* q_w    = (const float*)d_in[5];
  const float* q_bv   = (const float*)d_in[6];
  const float* kv_w   = (const float*)d_in[7];
  const float* kv_bv  = (const float*)d_in[8];
  const float* proj_w = (const float*)d_in[9];
  const float* proj_bv= (const float*)d_in[10];
  float* out = (float*)d_out;
  __bf16* ws = (__bf16*)d_ws;

  prep_weights<<<68, 256, 0, stream>>>(q_w, proj_w, kv_w, sr_w, ws);
  kv_path<<<BATCH, 256, 0, stream>>>(x, sr_b, ln_g, ln_b, kv_bv, ws, ws);
  swizzle_kv<<<64, 256, 0, stream>>>(ws);
  attention<<<BATCH * (NTOK / 128), 256, 0, stream>>>(x, q_bv, proj_bv, ws, out);
}
